// DiscriminatorLmk_47287589929916
// MI455X (gfx1250) — matrix operand order repaired and checked by reading, not compiled, not run on hardware
//
#include <hip/hip_runtime.h>
#include <hip/hip_bf16.h>
#include <math.h>

// ---------------------------------------------------------------------------
// Types for CDNA5 WMMA
// ---------------------------------------------------------------------------
typedef _Float16 f16t;
typedef __attribute__((ext_vector_type(16))) _Float16 v16h;
typedef __attribute__((ext_vector_type(8)))  _Float16 v8h;
typedef __attribute__((ext_vector_type(8)))  float    v8f;

union V16 { v16h v; v8h h[2]; };

#define WMMA_F16(a, b, c) \
  __builtin_amdgcn_wmma_f32_16x16x32_f16(false, (a), false, (b), (short)0, (c), false, false)

// Problem constants
#define Bsz   128
#define Lsz   1024
#define INRAW 136
#define INPAD 160
#define FF1   256
#define FF2   512
#define Hh    256
#define GN    1024          // 4*H
#define RB    16            // batch rows per LSTM block

// ---------------------------------------------------------------------------
// Global-address-space accessors: force global_load/global_store (not FLAT),
// even when the pointer was selected through a runtime-indexed array.
// ---------------------------------------------------------------------------
typedef __attribute__((address_space(1))) f16t  gf16;
typedef __attribute__((address_space(1))) float gf32;

static __device__ __forceinline__ v8h gload8(const f16t* p) {
  return *(const __attribute__((address_space(1))) v8h*)(unsigned long long)p;
}
static __device__ __forceinline__ float gloadf(const float* p) {
  return *(const gf32*)(unsigned long long)p;
}
static __device__ __forceinline__ void gstoreh(f16t* p, f16t v) {
  *(gf16*)(unsigned long long)p = v;
}
static __device__ __forceinline__ void gstoref(float* p, float v) {
  *(gf32*)(unsigned long long)p = v;
}

// ---------------------------------------------------------------------------
// Fragment loaders (layouts per CDNA5 ISA 7.12.2, wave32)
// A: 16x32 f16, row m = lane&15; lanes<16 hold K {0..7,16..23}, lanes>=16 K {8..15,24..31}
// B: 32x16 f16, col n = lane&15; lanes<16 hold K {0..7,16..23}, lanes>=16 K {8..15,24..31}
// C/D: 16x16 f32, col n = lane&15, row = (lane>>4)*8 + vgpr
// ---------------------------------------------------------------------------
static __device__ __forceinline__ v16h load_a_lds(const f16t* base, int hi) {
  V16 r;                                 // base = LDS row m at k-tile start
  r.h[0] = *(const v8h*)(base + hi * 8);
  r.h[1] = *(const v8h*)(base + 16 + hi * 8);
  return r.v;
}
static __device__ __forceinline__ v16h load_a_global(const f16t* base, int hi) {
  V16 r;
  r.h[0] = gload8(base + hi * 8);
  r.h[1] = gload8(base + 16 + hi * 8);
  return r.v;
}
static __device__ __forceinline__ v16h load_b_global(const f16t* wrow, int hi) {
  V16 r;
  r.h[0] = gload8(wrow + hi * 8);
  r.h[1] = gload8(wrow + 16 + hi * 8);
  return r.v;
}

// ---------------------------------------------------------------------------
// Utility kernels
// ---------------------------------------------------------------------------
__global__ void cvt_pad_kernel(const float* __restrict__ src, f16t* __restrict__ dst,
                               int R, int C, int Cp) {
  size_t i = (size_t)blockIdx.x * blockDim.x + threadIdx.x;
  size_t tot = (size_t)R * Cp;
  if (i >= tot) return;
  int r = (int)(i / Cp);
  int c = (int)(i % Cp);
  dst[i] = (c < C) ? (f16t)src[(size_t)r * C + c] : (f16t)0.0f;
}

__global__ void bias_combine_kernel(const float* __restrict__ a, const float* __restrict__ b,
                                    float* __restrict__ dst, int n) {
  int i = blockIdx.x * blockDim.x + threadIdx.x;
  if (i < n) dst[i] = a[i] + b[i];
}

__global__ void set_ones_col_kernel(f16t* a, int R, int Cp, int c) {
  int r = blockIdx.x * blockDim.x + threadIdx.x;
  if (r < R) a[(size_t)r * Cp + c] = (f16t)1.0f;
}
__global__ void set_bias_col_kernel(f16t* a, const float* b, int R, int Cp, int c) {
  int r = blockIdx.x * blockDim.x + threadIdx.x;
  if (r < R) a[(size_t)r * Cp + c] = (f16t)b[r];
}
__global__ void bias_relu_f16_kernel(f16t* a, const float* b, size_t rows, int N) {
  size_t i = (size_t)blockIdx.x * blockDim.x + threadIdx.x;
  if (i >= rows * (size_t)N) return;
  int n = (int)(i % N);
  float v = (float)a[i] + b[n];
  a[i] = (f16t)(v > 0.0f ? v : 0.0f);
}

// ---------------------------------------------------------------------------
// Generic WMMA GEMM:  C[M,N] = act(A[M,K] @ Bw[N,K]^T), f16 in / f16 out.
// Block = 128 threads (4 waves); wave tile = 16 x 64. grid = (M/16, N/256).
// ---------------------------------------------------------------------------
template <bool RELU>
__global__ __launch_bounds__(128)
void gemm_wmma_kernel(const f16t* __restrict__ A, const f16t* __restrict__ Bw,
                      f16t* __restrict__ C, int M, int N, int K) {
  const int wave = threadIdx.x >> 5;
  const int lane = threadIdx.x & 31;
  const int col  = lane & 15;
  const int hi   = lane >> 4;
  const int m0   = blockIdx.x * 16;
  const int n0   = blockIdx.y * 256 + wave * 64;

  v8f zero = {};
  v8f acc[4];
#pragma unroll
  for (int j = 0; j < 4; ++j) acc[j] = zero;

  const int ktiles = K >> 5;
  for (int kt = 0; kt < ktiles; ++kt) {
    v16h a = load_a_global(A + (size_t)(m0 + col) * K + kt * 32, hi);
#pragma unroll
    for (int j = 0; j < 4; ++j) {
      v16h b = load_b_global(Bw + (size_t)(n0 + 16 * j + col) * K + kt * 32, hi);
      acc[j] = WMMA_F16(a, b, acc[j]);
    }
  }

  const int mr = hi * 8;
#pragma unroll
  for (int j = 0; j < 4; ++j) {
#pragma unroll
    for (int r = 0; r < 8; ++r) {
      float v = acc[j][r];
      if (RELU) v = v > 0.0f ? v : 0.0f;
      gstoreh(C + (size_t)(m0 + mr + r) * N + n0 + 16 * j + col, (f16t)v);
    }
  }
}

// ---------------------------------------------------------------------------
// LSTM recurrent kernel helpers
// ---------------------------------------------------------------------------
static __device__ __forceinline__ void accum_gates(v8f acc[8], const f16t* W /*global [GN,Hh]*/,
                                                   const f16t* hL /*LDS [RB,Hh]*/,
                                                   int nw, int col, int hi) {
#pragma unroll
  for (int kt = 0; kt < 8; ++kt) {
    v16h a = load_a_lds(hL + (size_t)col * Hh + kt * 32, hi);
#pragma unroll
    for (int j = 0; j < 8; ++j) {
      v16h b = load_b_global(W + (size_t)(nw + 16 * j + col) * Hh + kt * 32, hi);
      acc[j] = WMMA_F16(a, b, acc[j]);
    }
  }
}

// Issue async global->LDS copy of one (row,seg) strip of the next xg0 slice.
// GVS addressing: saddr = uniform base (block,t), vaddr = per-lane byte offset.
static __device__ __forceinline__ void issue_xg_async(const f16t* base_uniform,
                                                      const f16t* lds_dst,
                                                      unsigned g_off_bytes) {
  unsigned lds_a = (unsigned)(unsigned long long)lds_dst;
  unsigned long long sbase = (unsigned long long)base_uniform;
#pragma unroll
  for (int q = 0; q < 8; ++q) {
    asm volatile("global_load_async_to_lds_b128 %0, %1, %2 offset:0"
                 :: "v"(lds_a + q * 16), "v"(g_off_bytes + q * 16), "s"(sbase)
                 : "memory");
  }
}

// ---------------------------------------------------------------------------
// Persistent 3-layer LSTM recurrence + decision head + ragged mean.
// grid = 8 blocks x 256 threads; block owns 16 batch rows for all 1024 steps.
// Wave w computes gate columns [128w,128w+128) via WMMA; h/c/gates in LDS;
// xg0 slices double-buffered in LDS via async loads (ASYNCcnt).
// ---------------------------------------------------------------------------
__global__ __launch_bounds__(256)
void lstm_kernel(const f16t* __restrict__ xg0,   // [B, L, 1024] f16
                 const f16t* __restrict__ whh0, const f16t* __restrict__ wih1,
                 const f16t* __restrict__ whh1, const f16t* __restrict__ wih2,
                 const f16t* __restrict__ whh2,
                 const float* __restrict__ bg0, const float* __restrict__ bg1,
                 const float* __restrict__ bg2,
                 const float* __restrict__ Wd, const float* __restrict__ bd,
                 const int* __restrict__ length, float* __restrict__ out) {
  extern __shared__ char smem[];
  f16t*  hS    = (f16t*)smem;                       // [3][RB][Hh] f16  : 24576 B
  float* cS    = (float*)(smem + 24576);            // [3][RB][Hh] f32  : 49152 B
  float* gates = (float*)(smem + 73728);            // [RB][GN]   f32  : 65536 B
  f16t*  xgS   = (f16t*)(smem + 139264);            // [2][RB][GN] f16 : 65536 B
  float* bgS   = (float*)(smem + 204800);           // [3][GN]    f32  : 12288 B
  float* WdS   = (float*)(smem + 217088);           // [Hh]       f32  : 1024 B
  float* red   = (float*)(smem + 218112);           // [RB][16]   f32  : 1024 B
  float* dsum  = (float*)(smem + 219136);           // [RB]       f32  : 64 B
  int*   lenS  = (int*)(smem + 219200);             // [RB]       i32  : 64 B

  const int tid  = threadIdx.x;
  const int wave = tid >> 5;
  const int lane = tid & 31;
  const int col  = lane & 15;
  const int hi   = lane >> 4;
  const int blk  = blockIdx.x;
  const int row  = tid >> 4;       // update-phase row  (0..15)
  const int seg  = tid & 15;       // update-phase segment

  // ---- one-time staging ----
  for (int i = tid; i < 3 * RB * Hh; i += 256) { hS[i] = (f16t)0.0f; cS[i] = 0.0f; }
  for (int i = tid; i < GN; i += 256) {
    bgS[i]          = gloadf(bg0 + i);
    bgS[GN + i]     = gloadf(bg1 + i);
    bgS[2 * GN + i] = gloadf(bg2 + i);
  }
  for (int i = tid; i < Hh; i += 256) WdS[i] = gloadf(Wd + i);
  if (tid < RB) { dsum[tid] = 0.0f; lenS[tid] = length[RB * blk + tid]; }
  const float bd0 = gloadf(bd);
  __syncthreads();

  const f16t* whh[3] = { whh0, whh1, whh2 };
  const f16t* wih[3] = { nullptr, wih1, wih2 };
  const int nw = wave * 128;

  // prologue: prefetch xg slice for t=0 into buffer 0
  {
    const f16t* baseT = xg0 + (size_t)(RB * blk) * Lsz * GN;   // + t*GN with t=0
    const f16t* dst   = xgS + (size_t)row * GN + seg * 64;     // buffer 0
    unsigned goff = (unsigned)row * (Lsz * GN * 2u) + (unsigned)seg * 128u;
    issue_xg_async(baseT, dst, goff);
  }

  for (int t = 0; t < Lsz; ++t) {
    for (int layer = 0; layer < 3; ++layer) {
      v8f zero = {};
      v8f acc[8];
#pragma unroll
      for (int j = 0; j < 8; ++j) acc[j] = zero;

      accum_gates(acc, whh[layer], hS + (size_t)layer * RB * Hh, nw, col, hi);
      if (layer > 0)
        accum_gates(acc, wih[layer], hS + (size_t)(layer - 1) * RB * Hh, nw, col, hi);

      // spill D fragments to the LDS gate buffer
#pragma unroll
      for (int j = 0; j < 8; ++j)
#pragma unroll
        for (int r = 0; r < 8; ++r)
          gates[(size_t)(hi * 8 + r) * GN + nw + 16 * j + col] = acc[j][r];

      if (layer == 0)  // xg slice for this step must have landed in LDS
        asm volatile("s_wait_asynccnt 0x0" ::: "memory");
      __syncthreads();

      // pointwise LSTM cell update: 256 threads x 16 hidden units each
      {
        const float* bgl = bgS + (size_t)layer * GN;
        const f16t*  xr  = xgS + ((size_t)(t & 1) * RB + row) * GN;  // layer 0 only
        float* crow = cS + ((size_t)layer * RB + row) * Hh;
        f16t*  hrow = hS + ((size_t)layer * RB + row) * Hh;
        const float* grow = gates + (size_t)row * GN;
        const int hbase = seg * 16;
#pragma unroll 4
        for (int u = 0; u < 16; ++u) {
          int n = hbase + u;
          float iv = grow[n]       + bgl[n];
          float fv = grow[n + 256] + bgl[n + 256];
          float gv = grow[n + 512] + bgl[n + 512];
          float ov = grow[n + 768] + bgl[n + 768];
          if (layer == 0) {
            iv += (float)xr[n];       fv += (float)xr[n + 256];
            gv += (float)xr[n + 512]; ov += (float)xr[n + 768];
          }
          float ig = 1.0f / (1.0f + __expf(-iv));
          float fg = 1.0f / (1.0f + __expf(-fv));
          float gg = tanhf(gv);
          float og = 1.0f / (1.0f + __expf(-ov));
          float cn = fg * crow[n] + ig * gg;
          crow[n] = cn;
          hrow[n] = (f16t)(og * tanhf(cn));
        }
      }
      __syncthreads();

      // after layer 0 consumed xg[t], start async fetch of xg[t+1]
      if (layer == 0 && t + 1 < Lsz) {
        const f16t* baseT = xg0 + ((size_t)(RB * blk) * Lsz + (size_t)(t + 1)) * GN;
        const f16t* dst   = xgS + ((size_t)((t + 1) & 1) * RB + row) * GN + seg * 64;
        unsigned goff = (unsigned)row * (Lsz * GN * 2u) + (unsigned)seg * 128u;
        issue_xg_async(baseT, dst, goff);
      }
    }

    // decision head: dot(h2[row], Wd) + bd, masked accumulation
    {
      const f16t* h2 = hS + ((size_t)2 * RB + row) * Hh;
      float p = 0.0f;
#pragma unroll
      for (int jj = 0; jj < 16; ++jj)
        p += (float)h2[seg * 16 + jj] * WdS[seg * 16 + jj];
      red[row * 16 + seg] = p;
      __syncthreads();
      if (tid < RB) {
        float s = bd0;
#pragma unroll
        for (int jj = 0; jj < 16; ++jj) s += red[tid * 16 + jj];
        if (t < lenS[tid]) dsum[tid] += s;
      }
      __syncthreads();
    }
  }

  if (tid < RB)
    gstoref(out + RB * blk + tid, dsum[tid] / (float)lenS[tid]);
}

// ---------------------------------------------------------------------------
// Host side
// ---------------------------------------------------------------------------
extern "C" void kernel_launch(void* const* d_in, const int* in_sizes, int n_in,
                              void* d_out, int out_size, void* d_ws, size_t ws_size,
                              hipStream_t stream) {
  (void)in_sizes; (void)n_in; (void)out_size; (void)ws_size;

  const float* x      = (const float*)d_in[0];
  const int*   length = (const int*)d_in[1];
  const float* W1     = (const float*)d_in[2];
  const float* b1     = (const float*)d_in[3];
  const float* W2     = (const float*)d_in[4];
  const float* b2     = (const float*)d_in[5];
  const float* wih0f  = (const float*)d_in[6];
  const float* whh0f  = (const float*)d_in[7];
  const float* bih0   = (const float*)d_in[8];
  const float* bhh0   = (const float*)d_in[9];
  const float* wih1f  = (const float*)d_in[10];
  const float* whh1f  = (const float*)d_in[11];
  const float* bih1   = (const float*)d_in[12];
  const float* bhh1   = (const float*)d_in[13];
  const float* wih2f  = (const float*)d_in[14];
  const float* whh2f  = (const float*)d_in[15];
  const float* bih2   = (const float*)d_in[16];
  const float* bhh2   = (const float*)d_in[17];
  const float* Wd     = (const float*)d_in[18];
  const float* bd     = (const float*)d_in[19];

  const size_t M = (size_t)Bsz * Lsz;   // 131072 tokens

  // --- carve workspace ---
  char* w = (char*)d_ws;
  size_t off = 0;
  auto carve = [&](size_t bytes) -> void* {
    void* p = w + off;
    off = (off + bytes + 255) & ~(size_t)255;
    return p;
  };
  f16t* xpad    = (f16t*)carve(M * INPAD * 2);
  f16t* w1p     = (f16t*)carve((size_t)FF1 * INPAD * 2);
  f16t* w2p     = (f16t*)carve((size_t)FF2 * FF1 * 2);
  f16t* wih0    = (f16t*)carve((size_t)GN * FF2 * 2);
  f16t* whh0    = (f16t*)carve((size_t)GN * Hh * 2);
  f16t* wih1    = (f16t*)carve((size_t)GN * Hh * 2);
  f16t* whh1    = (f16t*)carve((size_t)GN * Hh * 2);
  f16t* wih2    = (f16t*)carve((size_t)GN * Hh * 2);
  f16t* whh2    = (f16t*)carve((size_t)GN * Hh * 2);
  float* bg0    = (float*)carve(GN * 4);
  float* bg1    = (float*)carve(GN * 4);
  float* bg2    = (float*)carve(GN * 4);
  f16t* feat256 = (f16t*)carve(M * FF1 * 2);
  f16t* feat512 = (f16t*)carve(M * FF2 * 2);
  f16t* xg0     = (f16t*)carve(M * GN * 2);

  auto cvt = [&](const float* s, f16t* d, int R, int C, int Cp) {
    size_t tot = (size_t)R * Cp;
    int g = (int)((tot + 255) / 256);
    cvt_pad_kernel<<<g, 256, 0, stream>>>(s, d, R, C, Cp);
  };

  // convert/pad inputs & weights to f16 row-major
  cvt(x,     xpad, (int)M, INRAW, INPAD);
  cvt(W1,    w1p,  FF1, INRAW, INPAD);
  cvt(W2,    w2p,  FF2, FF1, FF1);
  cvt(wih0f, wih0, GN, FF2, FF2);
  cvt(whh0f, whh0, GN, Hh, Hh);
  cvt(wih1f, wih1, GN, Hh, Hh);
  cvt(whh1f, whh1, GN, Hh, Hh);
  cvt(wih2f, wih2, GN, Hh, Hh);
  cvt(whh2f, whh2, GN, Hh, Hh);

  bias_combine_kernel<<<4, 256, 0, stream>>>(bih0, bhh0, bg0, GN);
  bias_combine_kernel<<<4, 256, 0, stream>>>(bih1, bhh1, bg1, GN);
  bias_combine_kernel<<<4, 256, 0, stream>>>(bih2, bhh2, bg2, GN);

  // bake b1 into GEMM1 via a constant-1 column in the K padding region:
  // xpad[:, INRAW] = 1, w1p[:, INRAW] = b1  (rest of pad stays zero)
  set_ones_col_kernel<<<(int)((M + 255) / 256), 256, 0, stream>>>(xpad, (int)M, INPAD, INRAW);
  set_bias_col_kernel<<<1, 256, 0, stream>>>(w1p, b1, FF1, INPAD, INRAW);

  // GEMM1: feat256 = relu(xpad @ w1p^T + b1)
  {
    dim3 g((unsigned)(M / 16), FF1 / 256);
    gemm_wmma_kernel<true><<<g, 128, 0, stream>>>(xpad, w1p, feat256, (int)M, FF1, INPAD);
  }
  // GEMM2 (no act) then fused bias+ReLU pass: feat512 = relu(feat256 @ W2^T + b2)
  {
    dim3 g((unsigned)(M / 16), FF2 / 256);
    gemm_wmma_kernel<false><<<g, 128, 0, stream>>>(feat256, w2p, feat512, (int)M, FF2, FF1);
    size_t tot = M * FF2;
    bias_relu_f16_kernel<<<(int)((tot + 255) / 256), 256, 0, stream>>>(feat512, b2, M, FF2);
  }
  // GEMM3: xg0 = feat512 @ wih0^T  (layer-0 bias bg0 added in the cell update)
  {
    dim3 g((unsigned)(M / 16), GN / 256);
    gemm_wmma_kernel<false><<<g, 128, 0, stream>>>(feat512, wih0, xg0, (int)M, GN, FF2);
  }

  // recurrent LSTM + decision + ragged mean
  lstm_kernel<<<Bsz / RB, 256, 219264, stream>>>(
      xg0, whh0, wih1, whh1, wih2, whh2, bg0, bg1, bg2, Wd, bd, length, (float*)d_out);
}
